// VectorQuantizerEMA_57466662420695
// MI455X (gfx1250) — compile-verified, matrix operand order checked by reading, not run on hardware
//
#include <hip/hip_runtime.h>
#include <hip/hip_bf16.h>

// ---------------------------------------------------------------------------
// VQ-VAE vector quantizer for MI455X (gfx1250, wave32, WMMA + async-LDS).
//   z_e      : [N=65536, D=256] f32
//   codebook : [K=4096,  D=256] f32
// Outputs (concatenated in d_out, f32):
//   z_q_st [N*256], idx-as-float [N], loss [1]
// ---------------------------------------------------------------------------

#define DDIM   256
#define KCB    4096
#define TILE_K 32    // codebook rows staged per iteration (2 N-tiles)
#define WAVE_M 32    // rows per wave (2 M-tiles) -> 2x2 C tiles per wave
#define BLK_M  256   // 8 waves * 32 rows

typedef __attribute__((ext_vector_type(16))) __bf16 v16bf;
typedef __attribute__((ext_vector_type(8)))  __bf16 v8bf;
typedef __attribute__((ext_vector_type(8)))  float  v8f;
typedef __attribute__((ext_vector_type(8)))  unsigned short us8;

__device__ __forceinline__ unsigned short f32_to_bf16_rne(float f) {
    unsigned int u = __float_as_uint(f);
    unsigned int r = u + 0x7FFFu + ((u >> 16) & 1u);
    return (unsigned short)(r >> 16);
}

__device__ __forceinline__ v16bf cat8(v8bf lo, v8bf hi) {
    return __builtin_shufflevector(lo, hi,
        0, 1, 2, 3, 4, 5, 6, 7, 8, 9, 10, 11, 12, 13, 14, 15);
}

// Async stage: 32 rows x 256 bf16 = 16 KB, 256 threads x 4 b128 transfers.
// LDS writes bypass VGPRs; tracked by ASYNCcnt.
__device__ __forceinline__ void stage_async(unsigned short* dstBase,
                                            const unsigned short* src,
                                            int tid) {
    unsigned lds = (unsigned)(uintptr_t)dstBase + (unsigned)(tid * 64);
    unsigned long long g = (unsigned long long)(uintptr_t)(src + tid * 32);
    #pragma unroll
    for (int j = 0; j < 4; ++j) {
        unsigned l = lds + j * 16;
        unsigned long long ga = g + j * 16;
        asm volatile("global_load_async_to_lds_b128 %0, %1, off"
                     :: "v"(l), "v"(ga) : "memory");
    }
}

__device__ __forceinline__ void wait_async0() {
    asm volatile("s_wait_asynccnt 0" ::: "memory");
}

// ---------------------------------------------------------------------------
// Kernel 1: codebook f32 -> bf16, e_norm = ||e||^2, zero loss accumulator.
// ---------------------------------------------------------------------------
__global__ __launch_bounds__(256) void vq_prep_cb(
    const float* __restrict__ cb, unsigned short* __restrict__ cbbf,
    float* __restrict__ enorm, float* __restrict__ loss_accum) {
    __shared__ float red[256];
    const int n = blockIdx.x, d = threadIdx.x;
    const float c = cb[(size_t)n * DDIM + d];
    cbbf[(size_t)n * DDIM + d] = f32_to_bf16_rne(c);
    red[d] = c * c;
    __syncthreads();
    #pragma unroll
    for (int s = 128; s > 0; s >>= 1) {
        if (d < s) red[d] += red[d + s];
        __syncthreads();
    }
    if (d == 0) enorm[n] = red[0];
    if (n == 0 && d == 0) *loss_accum = 0.0f;
}

// ---------------------------------------------------------------------------
// Kernel 2: z f32 -> bf16, 8 elements/thread.
// ---------------------------------------------------------------------------
__global__ __launch_bounds__(256) void vq_convert_z(
    const float* __restrict__ z, unsigned short* __restrict__ zbf, int total) {
    const int i = (blockIdx.x * 256 + threadIdx.x) * 8;
    if (i >= total) return;
    const float4 f0 = *(const float4*)(z + i);
    const float4 f1 = *(const float4*)(z + i + 4);
    us8 o;
    o[0] = f32_to_bf16_rne(f0.x); o[1] = f32_to_bf16_rne(f0.y);
    o[2] = f32_to_bf16_rne(f0.z); o[3] = f32_to_bf16_rne(f0.w);
    o[4] = f32_to_bf16_rne(f1.x); o[5] = f32_to_bf16_rne(f1.y);
    o[6] = f32_to_bf16_rne(f1.z); o[7] = f32_to_bf16_rne(f1.w);
    *(us8*)(zbf + i) = o;
}

// ---------------------------------------------------------------------------
// Kernel 3: fused GEMM + argmin, double-buffered async LDS staging,
// 2x2 register blocking per wave.
// Wave owns rows [m0, m0+32): A tiles a0 (rows m0..m0+15) and a1 (+16) in
// registers (128 VGPRs). Per iteration 32 codebook rows are staged async into
// LDS; each B fragment feeds 2 WMMAs, each A fragment feeds 2 WMMAs:
//   c00 += a0*b0  c01 += a0*b1  c10 += a1*b0  c11 += a1*b1
// => 32 WMMAs per barrier, 512 B of LDS read per WMMA.
// C layout: VGPR j <-> M = j + 8*(lane>=16); N = lane&15 => k = k0 + (lane&15).
// ---------------------------------------------------------------------------
__global__ __launch_bounds__(256) void vq_argmin(
    const unsigned short* __restrict__ zbf,   // [N][256] bf16
    const unsigned short* __restrict__ cbbf,  // [K][256] bf16
    const float* __restrict__ enorm,          // [K]
    int* __restrict__ idx_out) {
    __shared__ __align__(16) unsigned short Bsh[2][TILE_K * DDIM];  // 2 x 16 KB

    const int tid  = threadIdx.x;
    const int wave = tid >> 5;
    const int lane = tid & 31;
    const int nlo  = lane & 15;
    const int hi   = lane >> 4;
    const int m0   = blockIdx.x * BLK_M + wave * WAVE_M;

    // ---- A fragments: a0 = rows m0..m0+15, a1 = rows m0+16..m0+31 ----
    v16bf a0[8], a1[8];
    #pragma unroll
    for (int dc = 0; dc < 8; ++dc) {
        const unsigned short* p0 = zbf + (size_t)(m0 + nlo) * DDIM + dc * 32 + hi * 8;
        const unsigned short* p1 = p0 + 16 * DDIM;
        a0[dc] = cat8(*(const v8bf*)p0, *(const v8bf*)(p0 + 16));
        a1[dc] = cat8(*(const v8bf*)p1, *(const v8bf*)(p1 + 16));
    }

    // minv/mini[j]    : M-tile 0 (rows m0..m0+15)
    // minv/mini[8+j]  : M-tile 1 (rows m0+16..m0+31)
    float minv[16];
    int   mini[16];
    #pragma unroll
    for (int j = 0; j < 16; ++j) { minv[j] = 3.4e38f; mini[j] = 0; }

    const int niter = KCB / TILE_K;  // 128

    // Prologue: stage tile 0 into buffer 0.
    stage_async(&Bsh[0][0], cbbf, tid);
    wait_async0();
    __syncthreads();

    for (int nk = 0; nk < niter; ++nk) {
        const int k0 = nk * TILE_K;
        if (nk + 1 < niter)
            stage_async(&Bsh[(nk + 1) & 1][0],
                        cbbf + (size_t)(k0 + TILE_K) * DDIM, tid);

        const unsigned short* buf = &Bsh[nk & 1][0];
        v8f c00 = {0.f,0.f,0.f,0.f,0.f,0.f,0.f,0.f};
        v8f c01 = {0.f,0.f,0.f,0.f,0.f,0.f,0.f,0.f};
        v8f c10 = {0.f,0.f,0.f,0.f,0.f,0.f,0.f,0.f};
        v8f c11 = {0.f,0.f,0.f,0.f,0.f,0.f,0.f,0.f};
        #pragma unroll
        for (int dc = 0; dc < 8; ++dc) {
            const unsigned short* bp0 = buf + nlo * DDIM + dc * 32 + hi * 8;
            const unsigned short* bp1 = bp0 + 16 * DDIM;
            v16bf b0 = cat8(*(const v8bf*)bp0, *(const v8bf*)(bp0 + 16));
            v16bf b1 = cat8(*(const v8bf*)bp1, *(const v8bf*)(bp1 + 16));
            c00 = __builtin_amdgcn_wmma_f32_16x16x32_bf16(
                      false, a0[dc], false, b0, (short)0, c00, false, false);
            c10 = __builtin_amdgcn_wmma_f32_16x16x32_bf16(
                      false, a1[dc], false, b0, (short)0, c10, false, false);
            c01 = __builtin_amdgcn_wmma_f32_16x16x32_bf16(
                      false, a0[dc], false, b1, (short)0, c01, false, false);
            c11 = __builtin_amdgcn_wmma_f32_16x16x32_bf16(
                      false, a1[dc], false, b1, (short)0, c11, false, false);
        }

        const float en0 = enorm[k0 + nlo];
        const float en1 = enorm[k0 + 16 + nlo];
        const int   ki0 = k0 + nlo;
        const int   ki1 = k0 + 16 + nlo;
        #pragma unroll
        for (int j = 0; j < 8; ++j) {
            float s = fmaf(-2.0f, c00[j], en0);
            bool lt = s < minv[j];
            mini[j] = lt ? ki0 : mini[j];
            minv[j] = lt ? s   : minv[j];
            s  = fmaf(-2.0f, c01[j], en1);
            lt = s < minv[j];
            mini[j] = lt ? ki1 : mini[j];
            minv[j] = lt ? s   : minv[j];
            s  = fmaf(-2.0f, c10[j], en0);
            lt = s < minv[8 + j];
            mini[8 + j] = lt ? ki0 : mini[8 + j];
            minv[8 + j] = lt ? s   : minv[8 + j];
            s  = fmaf(-2.0f, c11[j], en1);
            lt = s < minv[8 + j];
            mini[8 + j] = lt ? ki1 : mini[8 + j];
            minv[8 + j] = lt ? s   : minv[8 + j];
        }

        // Next tile fully landed in LDS + everyone done reading this buffer.
        wait_async0();
        __syncthreads();
    }

    // ---- argmin across the 16 lanes of each half-wave ----
    #pragma unroll
    for (int off = 8; off >= 1; off >>= 1) {
        #pragma unroll
        for (int j = 0; j < 16; ++j) {
            const float ov = __shfl_xor(minv[j], off, 32);
            const int   oi = __shfl_xor(mini[j], off, 32);
            const bool take = (ov < minv[j]) || (ov == minv[j] && oi < mini[j]);
            minv[j] = take ? ov : minv[j];
            mini[j] = take ? oi : mini[j];
        }
    }
    if (nlo == 0) {  // lanes 0 (M = j) and 16 (M = j+8) within each M-tile
        #pragma unroll
        for (int j = 0; j < 8; ++j) {
            idx_out[m0 + hi * 8 + j]      = mini[j];
            idx_out[m0 + 16 + hi * 8 + j] = mini[8 + j];
        }
    }
}

// ---------------------------------------------------------------------------
// Kernel 4: gather z_q = codebook[idx] (f32), write outputs, partial loss.
// 4 rows per block, float4 per thread (pure bandwidth).
// ---------------------------------------------------------------------------
__global__ __launch_bounds__(256) void vq_gather_loss(
    const float* __restrict__ z, const float* __restrict__ cb,
    const int* __restrict__ idx, float* __restrict__ out_zq,
    float* __restrict__ out_idxf, float* __restrict__ loss_accum) {
    __shared__ float red[256];
    const int t = threadIdx.x;
    const int r = t >> 6;        // row within block: 0..3
    const int c = t & 63;        // 64 threads per row, float4 each
    const int n = blockIdx.x * 4 + r;
    const int k = idx[n];
    const float4 q  = *(const float4*)(cb + (size_t)k * DDIM + c * 4);
    const float4 ze = *(const float4*)(z  + (size_t)n * DDIM + c * 4);
    *(float4*)(out_zq + (size_t)n * DDIM + c * 4) = q;
    const float d0 = q.x - ze.x, d1 = q.y - ze.y;
    const float d2 = q.z - ze.z, d3 = q.w - ze.w;
    red[t] = d0 * d0 + d1 * d1 + d2 * d2 + d3 * d3;
    __syncthreads();
    #pragma unroll
    for (int s = 32; s > 0; s >>= 1) {
        if (c < s) red[t] += red[t + s];
        __syncthreads();
    }
    if (c == 0) {
        atomicAdd(loss_accum, red[t]);
        out_idxf[n] = (float)k;
    }
}

__global__ void vq_finalize(const float* __restrict__ loss_accum,
                            float* __restrict__ out_loss, float scale) {
    *out_loss = (*loss_accum) * scale;
}

// ---------------------------------------------------------------------------
extern "C" void kernel_launch(void* const* d_in, const int* in_sizes, int n_in,
                              void* d_out, int out_size, void* d_ws, size_t ws_size,
                              hipStream_t stream) {
    const float* z  = (const float*)d_in[0];     // [N*256]
    const float* cb = (const float*)d_in[1];     // [K*256]
    const int N = in_sizes[0] / DDIM;            // 65536
    const int K = in_sizes[1] / DDIM;            // 4096
    (void)K; (void)n_in; (void)out_size; (void)ws_size;

    // Workspace layout (all offsets 256B-aligned):
    char* ws = (char*)d_ws;
    unsigned short* cbbf  = (unsigned short*)ws;                       // 2 MB
    float*          enorm = (float*)(ws + (size_t)2 * 1024 * 1024);    // 16 KB
    unsigned short* zbf   = (unsigned short*)(ws + (size_t)2 * 1024 * 1024 + 65536);
    int*            idxw  = (int*)((char*)zbf + (size_t)N * DDIM * 2);
    float*          lossw = (float*)((char*)idxw + (size_t)N * 4);

    float* out      = (float*)d_out;
    float* out_zq   = out;
    float* out_idxf = out + (size_t)N * DDIM;
    float* out_loss = out + (size_t)N * DDIM + N;

    vq_prep_cb    <<<KCB, 256, 0, stream>>>(cb, cbbf, enorm, lossw);
    vq_convert_z  <<<(N * DDIM) / (256 * 8), 256, 0, stream>>>(z, zbf, N * DDIM);
    vq_argmin     <<<N / BLK_M, 256, 0, stream>>>(zbf, cbbf, enorm, idxw);
    vq_gather_loss<<<N / 4, 256, 0, stream>>>(z, cb, idxw, out_zq, out_idxf, lossw);
    const float scale = 0.25f / (float)((size_t)N * DDIM);  // BETA * mean
    vq_finalize   <<<1, 1, 0, stream>>>(lossw, out_loss, scale);
}
